// ReflexSplit_24567212933939
// MI455X (gfx1250) — matrix operand order repaired and checked by reading, not compiled, not run on hardware
//
#include <hip/hip_runtime.h>

typedef __attribute__((ext_vector_type(16))) _Float16 v16h;
typedef __attribute__((ext_vector_type(8)))  _Float16 v8h;
typedef __attribute__((ext_vector_type(4)))  _Float16 v4h;
typedef __attribute__((ext_vector_type(8)))  float    v8f;

#define H_ 6
#define N_ 64
#define C_ 192
#define D_ 32

__device__ __forceinline__ v8f wmma16(v16h a, v16h b, v8f c) {
  // (neg_a, A, neg_b, B, c_mod, C, reuse_a, reuse_b)
  return __builtin_amdgcn_wmma_f32_16x16x32_f16(false, a, false, b, (short)0, c, false, false);
}

// A fragment: 16x32 f16 tile, source row-major (ld halfs). Two 16B vector loads.
// lane l: m = l&15 ; lanes 0-15 K={0..15}, lanes 16-31 K={16..31}
__device__ __forceinline__ v16h ldA(const _Float16* S, int ld, int mbase, int kbase, int lane) {
  const int m  = mbase + (lane & 15);
  const int g8 = (lane >> 4) << 4;
  const _Float16* row = S + m * ld + kbase + g8;
  v8h lo = *(const v8h*)(row);
  v8h hi = *(const v8h*)(row + 8);
  return __builtin_shufflevector(lo, hi, 0,1,2,3,4,5,6,7,8,9,10,11,12,13,14,15);
}

// B fragment from k-major LDS source (for P@V): B[k][n] = S[(kbase+k)*ld + nbase+n]
__device__ __forceinline__ v16h ldB(const _Float16* S, int ld, int kbase, int nbase, int lane) {
  v16h b;
  const int n   = nbase + (lane & 15);
  const int g16 = (lane >> 4) << 4;
  const _Float16* col = S + (kbase + g16) * ld + n;
#pragma unroll
  for (int i = 0; i < 16; ++i) b[i] = col[i * ld];
  return b;
}

// B fragment from n-major LDS source (for Q.K^T): contiguous 32B vector load
__device__ __forceinline__ v16h ldBT(const _Float16* S, int ld, int kbase, int nbase, int lane) {
  const _Float16* row = S + (nbase + (lane & 15)) * ld + kbase + ((lane >> 4) << 4);
  return *(const v16h*)row;
}

// B fragment from packed global weights: one 32B load per lane
__device__ __forceinline__ v16h ldBW(const _Float16* Wp, int NT, int kt, int nt, int lane) {
  return ((const v16h*)Wp)[((kt * NT + nt) << 5) | lane];
}

// ---------------- weight pack: fragment-ready order
// Wp[((kt*NT+nt)*32 + lane)*16 + e] = W[n*Kdim + k],  k = kt*32+((lane>>4)<<4)+e, n = nt*16+(lane&15)
__global__ void pack_w_kernel(const float* __restrict__ W, _Float16* __restrict__ Wp,
                              int Nout, int Kdim) {
  int i = blockIdx.x * blockDim.x + threadIdx.x;
  if (i >= Nout * Kdim) return;
  const int NT = Nout >> 4;
  const int e = i & 15;
  const int l = (i >> 4) & 31;
  const int tile = i >> 9;
  const int kt = tile / NT, nt = tile - kt * NT;
  const int k = kt * 32 + ((l >> 4) << 4) + e;
  const int n = nt * 16 + (l & 15);
  Wp[i] = (_Float16)W[n * Kdim + k];
}

// ---------------- differential attention over 6 heads, everything in LDS
__device__ __forceinline__ void diff_attention(
    const _Float16* qT, const _Float16* qR, int qld,
    const _Float16* kvT, const _Float16* kvR, int kvld, int koff0, int voff0,
    float lam, float scale, const float* __restrict__ rpb,
    float* S_T, float* S_R, _Float16* P_T, _Float16* P_R,
    _Float16* OUT_T, _Float16* OUT_R,
    int w, int lane, int tid)
{
  const int wp  = w >> 2;      // 0: T-side, 1: R-side
  const int wmt = w & 3;       // m-tile owned by this wave
  for (int h = 0; h < H_; ++h) {
    const int qoff = h * D_;
    const int koff = koff0 + h * D_;
    const int voff = voff0 + h * D_;
    // ---- scores: wave owns (p, mt) strip; Q fragment cached across 4 key tiles
    {
      const _Float16* Q  = wp ? qR  : qT;
      const _Float16* KV = wp ? kvR : kvT;
      float* S = wp ? S_R : S_T;
      const v16h aq = ldA(Q, qld, wmt * 16, qoff, lane);
      const int nl = lane & 15;
      const int mb = wmt * 16 + ((lane >> 4) << 3);
#pragma unroll
      for (int nt = 0; nt < 4; ++nt) {
        v8f acc = {};
        acc = wmma16(aq, ldBT(KV, kvld, koff, nt * 16, lane), acc);
        const int n  = nt * 16 + nl;
        const int kr = n >> 3, kc = n & 7;
#pragma unroll
        for (int r = 0; r < 8; ++r) {
          const int m = mb + r;
          const int idx = ((m >> 3) - kr + 7) * 15 + ((m & 7) - kc + 7); // rel-pos index
          S[m * N_ + n] = scale * acc[r] + rpb[idx * H_ + h];
        }
      }
    }
    __syncthreads();
    // ---- row softmax (128 rows over threads 0..127), float4 vectorized
    if (tid < 128) {
      float4* r4 = (float4*)(((tid & 64) ? S_R : S_T) + (tid & 63) * N_);
      float4 m4 = r4[0];
#pragma unroll 4
      for (int i = 1; i < 16; ++i) {
        float4 v = r4[i];
        m4.x = fmaxf(m4.x, v.x); m4.y = fmaxf(m4.y, v.y);
        m4.z = fmaxf(m4.z, v.z); m4.w = fmaxf(m4.w, v.w);
      }
      const float mx = fmaxf(fmaxf(m4.x, m4.y), fmaxf(m4.z, m4.w));
      float sum = 0.f;
#pragma unroll 4
      for (int i = 0; i < 16; ++i) {
        float4 v = r4[i];
        v.x = __expf(v.x - mx); v.y = __expf(v.y - mx);
        v.z = __expf(v.z - mx); v.w = __expf(v.w - mx);
        sum += (v.x + v.y) + (v.z + v.w);
        r4[i] = v;
      }
      const float inv = 1.f / sum;
#pragma unroll 4
      for (int i = 0; i < 16; ++i) {
        float4 v = r4[i];
        v.x *= inv; v.y *= inv; v.z *= inv; v.w *= inv;
        r4[i] = v;
      }
    }
    __syncthreads();
    // ---- differential combine (float4 in, v4h out)
    {
      const float4* at4 = (const float4*)S_T;
      const float4* ar4 = (const float4*)S_R;
      for (int i = tid; i < (N_ * N_) / 4; i += 256) {
        const float4 at = at4[i], ar = ar4[i];
        v4h pt, pr;
        pt[0] = (_Float16)(at.x - lam * ar.x); pr[0] = (_Float16)(ar.x - lam * at.x);
        pt[1] = (_Float16)(at.y - lam * ar.y); pr[1] = (_Float16)(ar.y - lam * at.y);
        pt[2] = (_Float16)(at.z - lam * ar.z); pr[2] = (_Float16)(ar.z - lam * at.z);
        pt[3] = (_Float16)(at.w - lam * ar.w); pr[3] = (_Float16)(ar.w - lam * at.w);
        *(v4h*)(P_T + 4 * i) = pt;
        *(v4h*)(P_R + 4 * i) = pr;
      }
    }
    __syncthreads();
    // ---- O = P @ V (64x32, K=64): both P fragments cached across the 2 n-tiles
    {
      const _Float16* P  = wp ? P_R : P_T;
      const _Float16* KV = wp ? kvR : kvT;
      _Float16* O = wp ? OUT_R : OUT_T;
      const v16h p0 = ldA(P, N_, wmt * 16, 0,  lane);
      const v16h p1 = ldA(P, N_, wmt * 16, 32, lane);
      const int n  = lane & 15;
      const int mb = wmt * 16 + ((lane >> 4) << 3);
#pragma unroll
      for (int nt = 0; nt < 2; ++nt) {
        v8f acc = {};
        acc = wmma16(p0, ldB(KV, kvld, 0,  voff + nt * 16, lane), acc);
        acc = wmma16(p1, ldB(KV, kvld, 32, voff + nt * 16, lane), acc);
#pragma unroll
        for (int r = 0; r < 8; ++r)
          O[(mb + r) * C_ + h * D_ + nt * 16 + n] = (_Float16)acc[r];
      }
    }
    __syncthreads();
  }
}

// ---------------- merged-head projection: O = A @ Wp^T + b -> global f32
// 8 strips (p, mt) = 8 waves; A fragments preloaded once per strip
__device__ __forceinline__ void project_out(const _Float16* A0, const _Float16* A1,
                                            const _Float16* Wp, const float* __restrict__ bias,
                                            float* __restrict__ O0, float* __restrict__ O1,
                                            int w, int lane)
{
  const int p = w >> 2, mt = w & 3;
  const _Float16* A = p ? A1 : A0;
  float* O = p ? O1 : O0;
  v16h af[6];
#pragma unroll
  for (int kt = 0; kt < 6; ++kt) af[kt] = ldA(A, C_, mt * 16, kt * 32, lane);
  const int nl = lane & 15;
  const int mb = mt * 16 + ((lane >> 4) << 3);
  for (int nt = 0; nt < 12; ++nt) {
    v8f acc = {};
#pragma unroll
    for (int kt = 0; kt < 6; ++kt)
      acc = wmma16(af[kt], ldBW(Wp, 12, kt, nt, lane), acc);
    const int n  = nt * 16 + nl;
    const float bb = bias[n];
#pragma unroll
    for (int r = 0; r < 8; ++r) O[(mb + r) * C_ + n] = acc[r] + bb;
  }
}

// LDS layout (bytes), 288 KB total with phase-based reuse
#define OFF_X16   0        // 24576  -> later S_T/S_R/P_T/P_R
#define OFF_Y16   24576
#define OFF_ST    0
#define OFF_SR    16384
#define OFF_PT    32768
#define OFF_PR    40960
#define OFF_TE    49152    // -> later OUT_T
#define OFF_RE    73728    // -> later OUT_R
#define OFF_TE2   98304
#define OFF_RE2   122880
#define OFF_QKVT  147456   // -> later Q_T2 / Q_R2 / KV_T / KV_R
#define OFF_QKVR  221184
#define OFF_QT2   147456
#define OFF_QR2   172032
#define OFF_KVT   196608
#define OFF_KVR   245760
#define LDS_BYTES 294912

__global__ __launch_bounds__(256, 1) void fused_diff_attn(
    const float* __restrict__ x, const float* __restrict__ y,
    const _Float16* __restrict__ w_sa_cr, const _Float16* __restrict__ w_sa_ct,
    const _Float16* __restrict__ w_ca_cr, const _Float16* __restrict__ w_ca_ct,
    const _Float16* __restrict__ w_sa_qkv, const float* __restrict__ sa_qkv_b,
    const _Float16* __restrict__ w_ca_q, const float* __restrict__ ca_q_b,
    const _Float16* __restrict__ w_ca_kv, const float* __restrict__ ca_kv_b,
    const _Float16* __restrict__ w_proj_sa, const float* __restrict__ proj_sa_b,
    const _Float16* __restrict__ w_proj_ca, const float* __restrict__ proj_ca_b,
    const float* __restrict__ rpb,
    const float* __restrict__ sa_enh_p, const float* __restrict__ ca_enh_p,
    const float* __restrict__ lam_sa_p, const float* __restrict__ lam_ca_p,
    float* __restrict__ out)
{
  __shared__ __align__(32) unsigned char smem[LDS_BYTES];
  const int b    = blockIdx.x;
  const int tid  = threadIdx.x;
  const int w    = tid >> 5;
  const int lane = tid & 31;
  const float scale = 0.1767766953f; // 1/sqrt(32)

  _Float16* X16  = (_Float16*)(smem + OFF_X16);
  _Float16* Y16  = (_Float16*)(smem + OFF_Y16);
  _Float16* TE   = (_Float16*)(smem + OFF_TE);
  _Float16* RE   = (_Float16*)(smem + OFF_RE);
  _Float16* TE2  = (_Float16*)(smem + OFF_TE2);
  _Float16* RE2  = (_Float16*)(smem + OFF_RE2);
  _Float16* QKVT = (_Float16*)(smem + OFF_QKVT);
  _Float16* QKVR = (_Float16*)(smem + OFF_QKVR);
  float*    S_T  = (float*)   (smem + OFF_ST);
  float*    S_R  = (float*)   (smem + OFF_SR);
  _Float16* P_T  = (_Float16*)(smem + OFF_PT);
  _Float16* P_R  = (_Float16*)(smem + OFF_PR);
  _Float16* OUT_T= (_Float16*)(smem + OFF_TE);
  _Float16* OUT_R= (_Float16*)(smem + OFF_RE);
  _Float16* Q_T2 = (_Float16*)(smem + OFF_QT2);
  _Float16* Q_R2 = (_Float16*)(smem + OFF_QR2);
  _Float16* KV_T = (_Float16*)(smem + OFF_KVT);
  _Float16* KV_R = (_Float16*)(smem + OFF_KVR);

  const float* xb = x + (size_t)b * N_ * C_;
  const float* yb = y + (size_t)b * N_ * C_;
  const float saE = *sa_enh_p, caE = *ca_enh_p;
  const float lsa = *lam_sa_p, lca = *lam_ca_p;
  const float lamSA = fminf(0.99f, fmaxf(0.01f, 1.f / (1.f + __expf(-lsa))));
  const float lamCA = fminf(0.99f, fmaxf(0.01f, 1.f / (1.f + __expf(-lca))));

  // ---- phase 1: stage x,y as f16 in LDS (float4 global loads, 8B LDS stores)
  {
    const float4* x4 = (const float4*)xb;
    const float4* y4 = (const float4*)yb;
    for (int i = tid; i < (N_ * C_) / 4; i += 256) {
      const float4 vx = x4[i], vy = y4[i];
      v4h hx, hy;
      hx[0] = (_Float16)vx.x; hx[1] = (_Float16)vx.y; hx[2] = (_Float16)vx.z; hx[3] = (_Float16)vx.w;
      hy[0] = (_Float16)vy.x; hy[1] = (_Float16)vy.y; hy[2] = (_Float16)vy.z; hy[3] = (_Float16)vy.w;
      *(v4h*)(X16 + 4 * i) = hx;
      *(v4h*)(Y16 + 4 * i) = hy;
    }
  }
  __syncthreads();

  // ---- phase 2: te / re / te2 / re2  (16 strips of (o, mt), 2 per wave)
  for (int s = w; s < 16; s += 8) {
    const int o = s >> 2, mt = s & 3;
    const bool rside = (o == 1 || o == 3);
    const _Float16* A = rside ? X16 : Y16;
    const _Float16* W = (o == 0) ? w_sa_cr : (o == 1) ? w_sa_ct : (o == 2) ? w_ca_cr : w_ca_ct;
    const _Float16* base16 = rside ? Y16 : X16;
    const float enh = (o < 2) ? saE : caE;
    _Float16* Dst = (o == 0) ? TE : (o == 1) ? RE : (o == 2) ? TE2 : RE2;
    v16h af[6];
#pragma unroll
    for (int kt = 0; kt < 6; ++kt) af[kt] = ldA(A, C_, mt * 16, kt * 32, lane);
    const int nl = lane & 15;
    const int mb = mt * 16 + ((lane >> 4) << 3);
    for (int nt = 0; nt < 12; ++nt) {
      v8f acc = {};
#pragma unroll
      for (int kt = 0; kt < 6; ++kt)
        acc = wmma16(af[kt], ldBW(W, 12, kt, nt, lane), acc);
      const int n = nt * 16 + nl;
      // batch all residual loads first (independent ds_loads, single wait),
      // then compute + store
      _Float16 res[8];
#pragma unroll
      for (int r = 0; r < 8; ++r) res[r] = base16[(mb + r) * C_ + n];
#pragma unroll
      for (int r = 0; r < 8; ++r)
        Dst[(mb + r) * C_ + n] = (_Float16)((float)res[r] + enh * acc[r]);
    }
  }
  __syncthreads();

  // ---- phase 3: SA QKV (8 strips of (p, mt) = 8 waves, 36 n-tiles each)
  {
    const int p = w >> 2, mt = w & 3;
    const _Float16* A = p ? RE : TE;
    _Float16* Dst = p ? QKVR : QKVT;
    v16h af[6];
#pragma unroll
    for (int kt = 0; kt < 6; ++kt) af[kt] = ldA(A, C_, mt * 16, kt * 32, lane);
    const int nl = lane & 15;
    const int mb = mt * 16 + ((lane >> 4) << 3);
    for (int nt = 0; nt < 36; ++nt) {
      v8f acc = {};
#pragma unroll
      for (int kt = 0; kt < 6; ++kt)
        acc = wmma16(af[kt], ldBW(w_sa_qkv, 36, kt, nt, lane), acc);
      const int n = nt * 16 + nl;
      const float bb = sa_qkv_b[n];
#pragma unroll
      for (int r = 0; r < 8; ++r) Dst[(mb + r) * (3 * C_) + n] = (_Float16)(acc[r] + bb);
    }
  }
  __syncthreads();

  // ---- phase 4: SA differential attention
  diff_attention(QKVT, QKVR, 3 * C_, QKVT, QKVR, 3 * C_, C_, 2 * C_,
                 lamSA, scale, rpb, S_T, S_R, P_T, P_R, OUT_T, OUT_R, w, lane, tid);

  // ---- phase 5: SA projection -> out[0][p*1024+b]
  {
    float* o_t = out + ((size_t)(0 * 2048 + 0 * 1024 + b)) * N_ * C_;
    float* o_r = out + ((size_t)(0 * 2048 + 1 * 1024 + b)) * N_ * C_;
    project_out(OUT_T, OUT_R, w_proj_sa, proj_sa_b, o_t, o_r, w, lane);
  }

  // ---- phase 6: CA Q and KV (16 strips, 2 per wave)
  for (int s = w; s < 16; s += 8) {
    const _Float16 *A, *W; _Float16* Dst; const float* bb;
    int mt, NT, dld, ntN;
    if (s < 8) {
      const int p = s >> 2; mt = s & 3;
      A = p ? RE2 : TE2; W = w_ca_q; bb = ca_q_b; Dst = p ? Q_R2 : Q_T2;
      NT = 12; dld = C_; ntN = 12;
    } else {
      const int ss = s - 8, p = ss >> 2; mt = ss & 3;
      A = p ? RE2 : TE2; W = w_ca_kv; bb = ca_kv_b; Dst = p ? KV_R : KV_T;
      NT = 24; dld = 2 * C_; ntN = 24;
    }
    v16h af[6];
#pragma unroll
    for (int kt = 0; kt < 6; ++kt) af[kt] = ldA(A, C_, mt * 16, kt * 32, lane);
    const int nl = lane & 15;
    const int mb = mt * 16 + ((lane >> 4) << 3);
    for (int nt = 0; nt < ntN; ++nt) {
      v8f acc = {};
#pragma unroll
      for (int kt = 0; kt < 6; ++kt)
        acc = wmma16(af[kt], ldBW(W, NT, kt, nt, lane), acc);
      const int n = nt * 16 + nl;
      const float bv = bb[n];
#pragma unroll
      for (int r = 0; r < 8; ++r) Dst[(mb + r) * dld + n] = (_Float16)(acc[r] + bv);
    }
  }
  __syncthreads();

  // ---- phase 7: CA differential attention
  diff_attention(Q_T2, Q_R2, C_, KV_T, KV_R, 2 * C_, 0, C_,
                 lamCA, scale, rpb, S_T, S_R, P_T, P_R, OUT_T, OUT_R, w, lane, tid);

  // ---- phase 8: CA projection -> out[1][p*1024+b]
  {
    float* o_t = out + ((size_t)(1 * 2048 + 0 * 1024 + b)) * N_ * C_;
    float* o_r = out + ((size_t)(1 * 2048 + 1 * 1024 + b)) * N_ * C_;
    project_out(OUT_T, OUT_R, w_proj_ca, proj_ca_b, o_t, o_r, w, lane);
  }
}

// workspace offsets (in _Float16 units)
#define W_SA_CR   0
#define W_SA_CT   36864
#define W_CA_CR   73728
#define W_CA_CT   110592
#define W_SA_QKV  147456
#define W_CA_Q    258048
#define W_CA_KV   294912
#define W_PROJ_SA 368640
#define W_PROJ_CA 405504

extern "C" void kernel_launch(void* const* d_in, const int* in_sizes, int n_in,
                              void* d_out, int out_size, void* d_ws, size_t ws_size,
                              hipStream_t stream) {
  const float* x         = (const float*)d_in[0];
  const float* y         = (const float*)d_in[1];
  const float* sa_qkv_w  = (const float*)d_in[2];
  const float* sa_qkv_b  = (const float*)d_in[3];
  const float* sa_ct_w   = (const float*)d_in[4];
  const float* sa_cr_w   = (const float*)d_in[5];
  const float* sa_enh    = (const float*)d_in[6];
  const float* ca_q_w    = (const float*)d_in[7];
  const float* ca_q_b    = (const float*)d_in[8];
  const float* ca_kv_w   = (const float*)d_in[9];
  const float* ca_kv_b   = (const float*)d_in[10];
  const float* ca_ct_w   = (const float*)d_in[11];
  const float* ca_cr_w   = (const float*)d_in[12];
  const float* ca_enh    = (const float*)d_in[13];
  const float* rpb_table = (const float*)d_in[14];
  const float* proj_sa_w = (const float*)d_in[15];
  const float* proj_sa_b = (const float*)d_in[16];
  const float* proj_ca_w = (const float*)d_in[17];
  const float* proj_ca_b = (const float*)d_in[18];
  const float* lambda_sa = (const float*)d_in[19];
  const float* lambda_ca = (const float*)d_in[20];

  _Float16* ws = (_Float16*)d_ws;

  auto pack = [&](const float* W, _Float16* Wp, int o, int i) {
    const int n = o * i;
    pack_w_kernel<<<(n + 255) / 256, 256, 0, stream>>>(W, Wp, o, i);
  };
  pack(sa_cr_w,   ws + W_SA_CR,   C_,     C_);
  pack(sa_ct_w,   ws + W_SA_CT,   C_,     C_);
  pack(ca_cr_w,   ws + W_CA_CR,   C_,     C_);
  pack(ca_ct_w,   ws + W_CA_CT,   C_,     C_);
  pack(sa_qkv_w,  ws + W_SA_QKV,  3 * C_, C_);
  pack(ca_q_w,    ws + W_CA_Q,    C_,     C_);
  pack(ca_kv_w,   ws + W_CA_KV,   2 * C_, C_);
  pack(proj_sa_w, ws + W_PROJ_SA, C_,     C_);
  pack(proj_ca_w, ws + W_PROJ_CA, C_,     C_);

  fused_diff_attn<<<1024, 256, 0, stream>>>(
      x, y,
      ws + W_SA_CR, ws + W_SA_CT, ws + W_CA_CR, ws + W_CA_CT,
      ws + W_SA_QKV, sa_qkv_b,
      ws + W_CA_Q, ca_q_b,
      ws + W_CA_KV, ca_kv_b,
      ws + W_PROJ_SA, proj_sa_b,
      ws + W_PROJ_CA, proj_ca_b,
      rpb_table, sa_enh, ca_enh, lambda_sa, lambda_ca,
      (float*)d_out);
  (void)in_sizes; (void)n_in; (void)out_size; (void)ws_size;
}
